// ProxE_16552803959285
// MI455X (gfx1250) — matrix operand order repaired and checked by reading, not compiled, not run on hardware
//
#include <hip/hip_runtime.h>
#include <hip/hip_bf16.h>
#include <math.h>

#define B_SZ   2048
#define D1     400
#define D2     400
#define OUT_C  96
#define FW     9
#define L_LEN  392          // D1 - FW + 1
#define FC_LEN 37632        // OUT_C * L_LEN (K of the big GEMM)
#define FC1_LEN 864
#define EPSF   1e-5f

#define NPAD_BIG 448        // fc_w rows padded to 7*64 for GEMM tiling
#define KPAD_X   416        // x1/x2 feature dim padded 400 -> 13*32

#define KCHUNK   128        // K depth staged per LDS buffer (4 WMMA substeps)
#define TILE_B   16384      // bytes per LDS B buffer: 4 substeps * 64 rows * 64B

typedef __bf16 bf16;
typedef bf16  v16bf __attribute__((ext_vector_type(16)));
typedef bf16  v8bf  __attribute__((ext_vector_type(8)));
typedef float v8f   __attribute__((ext_vector_type(8)));
typedef int   v4i   __attribute__((ext_vector_type(4)));

#define AS1 __attribute__((address_space(1)))
#define AS3 __attribute__((address_space(3)))
typedef AS1 v4i* gv4i_p;   // global-AS pointer to 4xint (async src)
typedef AS3 v4i* lv4i_p;   // LDS-AS pointer to 4xint (async dst)

// ---- async global->LDS availability (CDNA5 ASYNCcnt path) ----------------
#if defined(__has_builtin)
#  if __has_builtin(__builtin_amdgcn_global_load_async_to_lds_b128)
#    define HAVE_ASYNC_LDS 1
#  endif
#endif
#ifndef HAVE_ASYNC_LDS
#  define HAVE_ASYNC_LDS 0
#endif

__device__ __forceinline__ void wait_async(int pending) {
#if HAVE_ASYNC_LDS
#  if defined(__has_builtin) && __has_builtin(__builtin_amdgcn_s_wait_asynccnt)
  if (pending) __builtin_amdgcn_s_wait_asynccnt(4);
  else         __builtin_amdgcn_s_wait_asynccnt(0);
#  else
  if (pending) asm volatile("s_wait_asynccnt 4" ::: "memory");
  else         asm volatile("s_wait_asynccnt 0" ::: "memory");
#  endif
#else
  (void)pending;  // sync fallback: __syncthreads() orders the ds_stores
#endif
}

// ---- WMMA fragment loaders (bf16, wave32) --------------------------------
// A matrix 16x32 (MxK): lane L holds row (L&15); half = L>>4 selects
// K = {half*16 .. half*16+15} (contiguous per lane, matches B).
__device__ __forceinline__ v16bf load_fragA(const bf16* p) {
  v8bf lo = *(const v8bf*)(p);        // K block at +0
  v8bf hi = *(const v8bf*)(p + 8);    // K block at +8
  return __builtin_shufflevector(lo, hi, 0,1,2,3,4,5,6,7,8,9,10,11,12,13,14,15);
}
// B matrix 32x16 (KxN) from global: lane L holds col (L&15); lanes 0-15 hold
// K=0..15, lanes 16-31 hold K=16..31 (contiguous per lane).
__device__ __forceinline__ v16bf load_fragB(const bf16* p) {
  v8bf lo = *(const v8bf*)(p);
  v8bf hi = *(const v8bf*)(p + 8);
  return __builtin_shufflevector(lo, hi, 0,1,2,3,4,5,6,7,8,9,10,11,12,13,14,15);
}

// ---- LDS B-tile staging --------------------------------------------------
// Layout per buffer: [substep s:0..3][row n:0..63][chunk cs:0..3 * 16B],
// cs = c ^ ((n>>2)&3) XOR swizzle -> conflict-free fill and fragment reads.
__device__ __forceinline__ void stage_B(const bf16* __restrict__ Bm, int n0, int kb,
                                        uint4* smem, int bufOff, int tid) {
  int n  = tid >> 2;          // 0..63  : B row within tile
  int c  = tid & 3;           // 0..3   : 16B chunk (8 bf16) within 32-K substep
  int cs = c ^ ((n >> 2) & 3);
  const bf16* g = Bm + (size_t)(n0 + n) * FC_LEN + (size_t)kb + c * 8;
  char* l = (char*)smem + bufOff + n * 64 + cs * 16;
#pragma unroll
  for (int s = 0; s < 4; ++s) {
#if HAVE_ASYNC_LDS
    v4i* gsrc = (v4i*)(g + s * 32);        // reinterpret (+drop const), generic AS
    v4i* ldst = (v4i*)(l + s * 4096);      // generic AS, points into LDS
    __builtin_amdgcn_global_load_async_to_lds_b128(
        (gv4i_p)gsrc, (lv4i_p)ldst, 0, 0); // addrspacecast to AS1 / AS3
#else
    *(uint4*)(l + s * 4096) = *(const uint4*)(g + s * 32);
#endif
  }
}

// read one 32x16 B fragment (substep s, tile col-group row base) from LDS
__device__ __forceinline__ v16bf lds_fragB(const uint4* smem, int byteBase, int row, int half) {
  int sw = (row >> 2) & 3;
  const char* base = (const char*)smem + byteBase + row * 64;
  v8bf lo = *(const v8bf*)(base + (((2 * half)     ^ sw) * 16));
  v8bf hi = *(const v8bf*)(base + (((2 * half + 1) ^ sw) * 16));
  return __builtin_shufflevector(lo, hi, 0,1,2,3,4,5,6,7,8,9,10,11,12,13,14,15);
}

// ---- fc_w (400 x 37632 f32) -> bf16, padded to 448 rows ------------------
__global__ void k_cvt_fcw(const float* __restrict__ fcw, bf16* __restrict__ out) {
  size_t i = (size_t)blockIdx.x * blockDim.x + threadIdx.x;
  size_t total = (size_t)NPAD_BIG * FC_LEN;
  if (i >= total) return;
  size_t row = i / FC_LEN;
  out[i] = (row < (size_t)D1) ? (bf16)fcw[i] : (bf16)0.0f;
}

// ---- zero the K-pad columns (400..415) of x1bf/x2bf ----------------------
__global__ void k_zero_pad_x(bf16* __restrict__ x1, bf16* __restrict__ x2) {
  int i = blockIdx.x * blockDim.x + threadIdx.x;
  if (i >= B_SZ * 16) return;
  int row = i >> 4, c = D1 + (i & 15);
  x1[(size_t)row * KPAD_X + c] = (bf16)0.0f;
  x2[(size_t)row * KPAD_X + c] = (bf16)0.0f;
}

// ---- per-sample filters: k = R_w[r_idx] @ fc1_w.T + fc1_b ----------------
__global__ void k_fc1(const int* __restrict__ ridx, const float* __restrict__ Rw,
                      const float* __restrict__ w, const float* __restrict__ bias,
                      float* __restrict__ kout) {
  __shared__ float rs[D2];
  int b = blockIdx.x, tid = threadIdx.x;
  const float* rrow = Rw + (size_t)ridx[b] * D2;
  for (int d = tid; d < D2; d += blockDim.x) rs[d] = rrow[d];
  __syncthreads();
  for (int j = tid; j < FC1_LEN; j += blockDim.x) {
    const float* wr = w + (size_t)j * D2;
    float acc = bias[j];
#pragma unroll 4
    for (int d = 0; d < D2; ++d) acc = fmaf(rs[d], wr[d], acc);
    kout[(size_t)b * FC1_LEN + j] = acc;
  }
}

// ---- fused bn0 + sliding-window conv + bn1, emit flat in bf16 ------------
__global__ void k_conv(const int* __restrict__ eidx, const float* __restrict__ Ew,
                       const float* __restrict__ kbuf,
                       const float* __restrict__ bn0g, const float* __restrict__ bn0b,
                       const float* __restrict__ bn0m, const float* __restrict__ bn0v,
                       const float* __restrict__ bn1g, const float* __restrict__ bn1b,
                       const float* __restrict__ bn1m, const float* __restrict__ bn1v,
                       bf16* __restrict__ flat) {
  __shared__ float xs[D1];
  __shared__ float ks[FC1_LEN];
  __shared__ float s1[OUT_C], c1[OUT_C];
  int b = blockIdx.x, tid = threadIdx.x;
  float s0 = bn0g[0] * rsqrtf(bn0v[0] + EPSF);
  float c0 = bn0b[0] - bn0m[0] * s0;
  const float* er = Ew + (size_t)eidx[b] * D1;
  for (int d = tid; d < D1; d += blockDim.x) xs[d] = er[d] * s0 + c0;
  const float* kr = kbuf + (size_t)b * FC1_LEN;
  for (int j = tid; j < FC1_LEN; j += blockDim.x) ks[j] = kr[j];
  for (int o = tid; o < OUT_C; o += blockDim.x) {
    float s = bn1g[o] * rsqrtf(bn1v[o] + EPSF);
    s1[o] = s; c1[o] = bn1b[o] - bn1m[o] * s;
  }
  __syncthreads();
  bf16* orow = flat + (size_t)b * FC_LEN;
  for (int i = tid; i < FC_LEN; i += blockDim.x) {
    int o = i / L_LEN, l = i - o * L_LEN;
    const float* kk = ks + o * FW;
    float acc = 0.f;
#pragma unroll
    for (int f = 0; f < FW; ++f) acc = fmaf(xs[l + f], kk[f], acc);
    orow[i] = (bf16)(acc * s1[o] + c1[o]);
  }
}

// ---- big GEMM: (2048 x 37632 bf16) x fc_w_bf16^T, fused fc_b+bn2+ReLU ----
// Block = 8 waves = 128M x 64N. B tile shared via double-buffered LDS with
// async global->LDS fill; A fragments for the current chunk loaded into registers.
__global__ void k_gemm_big(const bf16* __restrict__ A, const bf16* __restrict__ Bm,
                           const float* __restrict__ fcb,
                           const float* __restrict__ bn2g, const float* __restrict__ bn2b,
                           const float* __restrict__ bn2m, const float* __restrict__ bn2v,
                           bf16* __restrict__ xout) {
  __shared__ uint4 smem[2 * TILE_B / 16];   // 32 KB
  int tid  = threadIdx.x;
  int lane = tid & 31;
  int wave = tid >> 5;
  int m0 = (blockIdx.y * 8 + wave) * 16;
  int n0 = blockIdx.x * 64;
  int sub  = lane & 15;
  int half = lane >> 4;

  const bf16* ap = A + (size_t)(m0 + sub) * FC_LEN + half * 16;

  v8f zero = {};
  v8f acc[4];
#pragma unroll
  for (int t = 0; t < 4; ++t) acc[t] = zero;

  stage_B(Bm, n0, 0, smem, 0, tid);         // prologue fill, buffer 0

  const int NCH = FC_LEN / KCHUNK;          // 294
  for (int ch = 0; ch < NCH; ++ch) {
    int kb = ch * KCHUNK;
    __syncthreads();                        // all reads of buf[(ch+1)&1] done
    int more = (ch + 1 < NCH);
    if (more) stage_B(Bm, n0, kb + KCHUNK, smem, ((ch + 1) & 1) * TILE_B, tid);
    wait_async(more);                       // current buffer's fill complete
    __syncthreads();                        // ... in every wave
    int bufOff = (ch & 1) * TILE_B;

    v16bf a[4];
#pragma unroll
    for (int s = 0; s < 4; ++s) a[s] = load_fragA(ap + kb + s * 32);
#pragma unroll
    for (int s = 0; s < 4; ++s) {
#pragma unroll
      for (int t = 0; t < 4; ++t) {
        v16bf bfr = lds_fragB(smem, bufOff + s * 4096, t * 16 + sub, half);
        acc[t] = __builtin_amdgcn_wmma_f32_16x16x32_bf16(
                   false, a[s], false, bfr, (short)0, acc[t], false, false);
      }
    }
  }

#pragma unroll
  for (int t = 0; t < 4; ++t) {
    int n = n0 + t * 16 + sub;
    if (n < D1) {
      float s2  = bn2g[n] * rsqrtf(bn2v[n] + EPSF);
      float add = fcb[n] - bn2m[n];
      float c2  = bn2b[n];
#pragma unroll
      for (int r = 0; r < 8; ++r) {
        float v = (acc[t][r] + add) * s2 + c2;
        v = fmaxf(v, 0.f);
        xout[(size_t)(m0 + half * 8 + r) * KPAD_X + n] = (bf16)v;
      }
    }
  }
}

// ---- logits GEMM: x1(2048x416) x x2^T + b_bias -> f32 (2048x2048) --------
__global__ void k_gemm_logits(const bf16* __restrict__ A, const bf16* __restrict__ Bm,
                              const float* __restrict__ bias, float* __restrict__ out) {
  int lane = threadIdx.x & 31;
  int wave = threadIdx.x >> 5;
  int m0 = (blockIdx.y * 8 + wave) * 16;
  int n0 = blockIdx.x * 64;
  int sub  = lane & 15;
  int half = lane >> 4;

  const bf16* ap = A  + (size_t)(m0 + sub) * KPAD_X + half * 16;
  const bf16* bp = Bm + (size_t)(n0 + sub) * KPAD_X + half * 16;

  v8f zero = {};
  v8f acc[4];
#pragma unroll
  for (int t = 0; t < 4; ++t) acc[t] = zero;

#pragma unroll
  for (int k = 0; k < KPAD_X; k += 32) {
    v16bf a = load_fragA(ap + k);
#pragma unroll
    for (int t = 0; t < 4; ++t) {
      v16bf bfr = load_fragB(bp + (size_t)t * 16 * KPAD_X + k);
      acc[t] = __builtin_amdgcn_wmma_f32_16x16x32_bf16(
                 false, a, false, bfr, (short)0, acc[t], false, false);
    }
  }

#pragma unroll
  for (int t = 0; t < 4; ++t) {
    int n = n0 + t * 16 + sub;
    float bb = bias[n];
#pragma unroll
    for (int r = 0; r < 8; ++r)
      out[(size_t)(m0 + half * 8 + r) * B_SZ + n] = acc[t][r] + bb;
  }
}

// ---- workspace layout (bytes, 256-aligned) -------------------------------
#define OFF_FCW  ((size_t)0)                                  // 448*37632*2 = 33,718,272
#define OFF_FLAT ((size_t)33718272)                           // 2048*37632*2 = 154,140,672
#define OFF_K    ((size_t)(33718272 + 154140672))             // 2048*864*4 = 7,077,888
#define OFF_X1   ((size_t)(33718272 + 154140672 + 7077888))   // 2048*416*2 = 1,703,936
#define OFF_X2   ((size_t)(OFF_X1 + 1703936))

extern "C" void kernel_launch(void* const* d_in, const int* in_sizes, int n_in,
                              void* d_out, int out_size, void* d_ws, size_t ws_size,
                              hipStream_t stream) {
  const int*   e1   = (const int*)d_in[0];
  const int*   r1   = (const int*)d_in[1];
  const int*   r2   = (const int*)d_in[2];
  const int*   e2   = (const int*)d_in[3];
  const float* Ew   = (const float*)d_in[4];
  const float* Rw   = (const float*)d_in[5];
  const float* fc1w = (const float*)d_in[6];
  const float* fc1b = (const float*)d_in[7];
  const float* fcw  = (const float*)d_in[8];
  const float* fcb  = (const float*)d_in[9];
  const float* bn0g = (const float*)d_in[10];
  const float* bn0b = (const float*)d_in[11];
  const float* bn0m = (const float*)d_in[12];
  const float* bn0v = (const float*)d_in[13];
  const float* bn1g = (const float*)d_in[14];
  const float* bn1b = (const float*)d_in[15];
  const float* bn1m = (const float*)d_in[16];
  const float* bn1v = (const float*)d_in[17];
  const float* bn2g = (const float*)d_in[18];
  const float* bn2b = (const float*)d_in[19];
  const float* bn2m = (const float*)d_in[20];
  const float* bn2v = (const float*)d_in[21];
  const float* bbias= (const float*)d_in[22];
  float* out = (float*)d_out;

  char* ws = (char*)d_ws;
  bf16*  fcw_bf = (bf16*)(ws + OFF_FCW);
  bf16*  flat   = (bf16*)(ws + OFF_FLAT);
  float* kbuf   = (float*)(ws + OFF_K);
  bf16*  x1bf   = (bf16*)(ws + OFF_X1);
  bf16*  x2bf   = (bf16*)(ws + OFF_X2);

  k_cvt_fcw<<<(NPAD_BIG * FC_LEN + 255) / 256, 256, 0, stream>>>(fcw, fcw_bf);
  k_zero_pad_x<<<(B_SZ * 16 + 255) / 256, 256, 0, stream>>>(x1bf, x2bf);

  // branch 1: (e1_idx, r_idx) -> x1
  k_fc1 <<<B_SZ, 256, 0, stream>>>(r1, Rw, fc1w, fc1b, kbuf);
  k_conv<<<B_SZ, 256, 0, stream>>>(e1, Ew, kbuf,
                                   bn0g, bn0b, bn0m, bn0v,
                                   bn1g, bn1b, bn1m, bn1v, flat);
  k_gemm_big<<<dim3(NPAD_BIG / 64, B_SZ / 128), 256, 0, stream>>>(
      flat, fcw_bf, fcb, bn2g, bn2b, bn2m, bn2v, x1bf);

  // branch 2: (e2_idx, r2_idx) -> x2
  k_fc1 <<<B_SZ, 256, 0, stream>>>(r2, Rw, fc1w, fc1b, kbuf);
  k_conv<<<B_SZ, 256, 0, stream>>>(e2, Ew, kbuf,
                                   bn0g, bn0b, bn0m, bn0v,
                                   bn1g, bn1b, bn1m, bn1v, flat);
  k_gemm_big<<<dim3(NPAD_BIG / 64, B_SZ / 128), 256, 0, stream>>>(
      flat, fcw_bf, fcb, bn2g, bn2b, bn2m, bn2v, x2bf);

  // logits = x1 @ x2.T + b_bias
  k_gemm_logits<<<dim3(B_SZ / 64, B_SZ / 128), 256, 0, stream>>>(x1bf, x2bf, bbias, out);

  (void)in_sizes; (void)n_in; (void)out_size; (void)ws_size;
}
